// SimplePGATLayer_57286273794129
// MI455X (gfx1250) — matrix operand order repaired and checked by reading, not compiled, not run on hardware
//
#include <hip/hip_runtime.h>
#include <hip/hip_bf16.h>

#define NN 50000
#define EE 800000
#define HH 4
#define CGAT 32
#define COBS 16
#define KOBS 2

typedef __bf16 bf16_t;
typedef bf16_t v16bf __attribute__((ext_vector_type(16)));
typedef bf16_t v8bf  __attribute__((ext_vector_type(8)));
typedef float  v8f   __attribute__((ext_vector_type(8)));

__device__ __forceinline__ bf16_t f2bf(float f) {
    union { float f; unsigned u; } v; v.f = f;
    unsigned r = v.u + 0x7FFFu + ((v.u >> 16) & 1u);   // round-to-nearest-even
    unsigned short hs = (unsigned short)(r >> 16);
    bf16_t b;
    __builtin_memcpy(&b, &hs, 2);
    return b;
}

// Elementwise f32 -> bf16 pack, 4 elements/thread (n % 4 == 0 for all uses).
__global__ void pack_bf16(const float* __restrict__ in, bf16_t* __restrict__ out, int n) {
    int i4 = (blockIdx.x * blockDim.x + threadIdx.x) * 4;
    if (i4 >= n) return;
    float4 v = *(const float4*)(in + i4);
    out[i4 + 0] = f2bf(v.x);
    out[i4 + 1] = f2bf(v.y);
    out[i4 + 2] = f2bf(v.z);
    out[i4 + 3] = f2bf(v.w);
}

// C[M,Nc] = A[M,Ka] * W[Nc,Ka]^T (+bias, +relu). A, W pre-packed bf16 row-major.
// grid.x = M/16, block = 128 (4 waves); wave w handles column tiles w, w+4, ...
// Requires M % 16 == 0; Ka % 32 == 0; Nc % 16 == 0 (compile-time).
template <int Ka, int Nc>
__global__ void gemm_bf16(const bf16_t* __restrict__ A, const bf16_t* __restrict__ W,
                          float* __restrict__ C, const float* __restrict__ bias, int relu) {
    const int lane = threadIdx.x & 31;
    const int wave = threadIdx.x >> 5;
    const int m0   = blockIdx.x << 4;
    const int lrow = lane & 15;
    const int hi   = lane >> 4;            // 0: lanes 0-15, 1: lanes 16-31
    const int row  = m0 + lrow;

    #pragma unroll
    for (int ct = wave; ct < (Nc >> 4); ct += 4) {
        const int n0   = ct << 4;
        const int ncol = n0 + lrow;
        v8f acc = {};
        #pragma unroll
        for (int kt = 0; kt < Ka; kt += 32) {
            // A fragment: row `row`, K = kt + hi*16 + {0..15} (contiguous, 32B aligned)
            v16bf a = *(const v16bf*)(A + (size_t)row * Ka + kt + hi * 16);
            // B fragment: col `ncol`, K = kt + hi*16 + {0..15} (contiguous, 32B aligned)
            v16bf b = *(const v16bf*)(W + (size_t)ncol * Ka + kt + hi * 16);
            acc = __builtin_amdgcn_wmma_f32_16x16x32_bf16(
                      false, a, false, b, (short)0, acc, false, false);
        }
        float bv = bias ? bias[ncol] : 0.0f;
        #pragma unroll
        for (int v = 0; v < 8; ++v) {
            int r = m0 + v + hi * 8;       // D: lanes 16-31 hold M = vgpr + 8
            float val = acc[v] + bv;
            if (relu) val = fmaxf(val, 0.0f);
            C[(size_t)r * Nc + n0 + lrow] = val;
        }
    }
}

// a_src[n,h] = <xh[n,h,:], att_src[h,:]>, same for a_dst. One thread per (n,h).
__global__ void k_attn_coeff(const float* __restrict__ xh,
                             const float* __restrict__ att_src,
                             const float* __restrict__ att_dst,
                             float* __restrict__ asrc, float* __restrict__ adst) {
    int idx = blockIdx.x * blockDim.x + threadIdx.x;
    if (idx >= NN * HH) return;
    int n = idx >> 2, h = idx & 3;
    const float* xr = xh + (size_t)n * 128 + h * CGAT;
    const float* ws = att_src + h * CGAT;
    const float* wd = att_dst + h * CGAT;
    float s1 = 0.f, s2 = 0.f;
    #pragma unroll
    for (int c = 0; c < CGAT; ++c) { s1 += xr[c] * ws[c]; s2 += xr[c] * wd[c]; }
    asrc[idx] = s1;
    adst[idx] = s2;
}

// Pass 1: e = leaky_relu(a_src[src]+a_dst[dst]); store e; atomic segment-max via
// order-preserving u32 key (f>=0 -> bits|0x80000000, f<0 -> ~bits).
__global__ void k_edge_max(const int* __restrict__ ei,
                           const float* __restrict__ asrc, const float* __restrict__ adst,
                           float* __restrict__ ebuf, unsigned int* __restrict__ mkey) {
    int e = blockIdx.x * blockDim.x + threadIdx.x;
    if (e >= EE) return;
    int s = ei[e];
    int d = ei[EE + e];
    #pragma unroll
    for (int h = 0; h < HH; ++h) {
        float v = asrc[s * HH + h] + adst[d * HH + h];
        v = (v > 0.f) ? v : 0.2f * v;
        ebuf[(size_t)e * HH + h] = v;
        unsigned bits = __float_as_uint(v);
        unsigned key  = (bits & 0x80000000u) ? ~bits : (bits | 0x80000000u);
        atomicMax(&mkey[d * HH + h], key);
    }
}

// Decode max keys to float in place; empty segments (key==0) -> 0 (matches reference).
__global__ void k_decode_max(unsigned int* __restrict__ mkey) {
    int i = blockIdx.x * blockDim.x + threadIdx.x;
    if (i >= NN * HH) return;
    unsigned key = mkey[i];
    float m = 0.0f;
    if (key != 0u) {
        unsigned bits = (key & 0x80000000u) ? (key ^ 0x80000000u) : ~key;
        m = __uint_as_float(bits);
    }
    ((float*)mkey)[i] = m;
}

// Pass 2: ex = exp(e - m[dst]); store ex in place; atomic-add denominators.
__global__ void k_edge_exp(const int* __restrict__ ei, float* __restrict__ ebuf,
                           const float* __restrict__ mval, float* __restrict__ den) {
    int e = blockIdx.x * blockDim.x + threadIdx.x;
    if (e >= EE) return;
    int d = ei[EE + e];
    #pragma unroll
    for (int h = 0; h < HH; ++h) {
        float ex = __expf(ebuf[(size_t)e * HH + h] - mval[d * HH + h]);
        ebuf[(size_t)e * HH + h] = ex;
        atomicAdd(&den[d * HH + h], ex);
    }
}

// Pass 3: one wave per edge. alpha[h] = ex[h]/max(den,1e-16); lane j scatters
// channels {j, j+32, j+64, j+96} (head = c/32) into combined[dst, 0:128].
__global__ void k_edge_aggregate(const int* __restrict__ ei, const float* __restrict__ ebuf,
                                 const float* __restrict__ den, const float* __restrict__ xh,
                                 float* __restrict__ comb) {
    int widx = (blockIdx.x * blockDim.x + threadIdx.x) >> 5;
    int lane = threadIdx.x & 31;
    if (widx >= EE) return;
    int s = ei[widx];
    int d = ei[EE + widx];
    float alpha[HH];
    #pragma unroll
    for (int h = 0; h < HH; ++h)
        alpha[h] = ebuf[(size_t)widx * HH + h] / fmaxf(den[d * HH + h], 1e-16f);
    const float* xs = xh + (size_t)s * 128;
    float* cd = comb + (size_t)d * 192;
    #pragma unroll
    for (int j = 0; j < 4; ++j) {
        int c = lane + 32 * j;          // head = j
        atomicAdd(&cd[c], alpha[j] * xs[c]);
    }
}

// Obstacle attention epilogue + gat_bias fold. One thread per (n,h).
__global__ void k_obstacle(const float* __restrict__ Qp, const float* __restrict__ KP,
                           const float* __restrict__ VP, const float* __restrict__ apos,
                           const float* __restrict__ opos, const float* __restrict__ cdecay,
                           const float* __restrict__ gat_bias, float* __restrict__ comb) {
    int idx = blockIdx.x * blockDim.x + threadIdx.x;
    if (idx >= NN * HH) return;
    int n = idx >> 2, h = idx & 3;
    float cd = cdecay[0];

    float qv[COBS];
    const float* qr = Qp + (size_t)n * 64 + h * COBS;
    #pragma unroll
    for (int c = 0; c < COBS; ++c) qv[c] = qr[c];

    float ax = apos[n * 2 + 0], ay = apos[n * 2 + 1];
    float sc[KOBS];
    #pragma unroll
    for (int k = 0; k < KOBS; ++k) {
        const float* kr = KP + ((size_t)n * KOBS + k) * 64 + h * COBS;
        float s = 0.f;
        #pragma unroll
        for (int c = 0; c < COBS; ++c) s += qv[c] * kr[c];
        float dx = ax - opos[(n * KOBS + k) * 2 + 0];
        float dy = ay - opos[(n * KOBS + k) * 2 + 1];
        float dist = sqrtf(dx * dx + dy * dy);
        sc[k] = s * __expf(-cd * dist);
    }
    float mx = fmaxf(sc[0], sc[1]);
    float e0 = __expf(sc[0] - mx), e1 = __expf(sc[1] - mx);
    float inv = 1.0f / (e0 + e1);
    float a0 = e0 * inv, a1 = e1 * inv;

    const float* v0 = VP + ((size_t)n * KOBS + 0) * 64 + h * COBS;
    const float* v1 = VP + ((size_t)n * KOBS + 1) * 64 + h * COBS;
    float* co = comb + (size_t)n * 192 + 128 + h * COBS;
    #pragma unroll
    for (int c = 0; c < COBS; ++c) co[c] = a0 * v0[c] + a1 * v1[c];

    float* ca = comb + (size_t)n * 192 + h * CGAT;
    const float* gb = gat_bias + h * CGAT;
    #pragma unroll
    for (int c = 0; c < CGAT; ++c) ca[c] += gb[c];
}

static inline int pk_blocks(int n) { return (n / 4 + 255) / 256; }

extern "C" void kernel_launch(void* const* d_in, const int* in_sizes, int n_in,
                              void* d_out, int out_size, void* d_ws, size_t ws_size,
                              hipStream_t stream) {
    const float* x        = (const float*)d_in[0];
    const int*   ei       = (const int*)  d_in[1];
    const float* qf       = (const float*)d_in[2];
    const float* obsf     = (const float*)d_in[3];
    const float* apos     = (const float*)d_in[4];
    const float* opos     = (const float*)d_in[5];
    const float* gat_w    = (const float*)d_in[6];
    const float* att_src  = (const float*)d_in[7];
    const float* att_dst  = (const float*)d_in[8];
    const float* gat_bias = (const float*)d_in[9];
    const float* wq       = (const float*)d_in[10];
    const float* wk       = (const float*)d_in[11];
    const float* wv       = (const float*)d_in[12];
    const float* cdecay   = (const float*)d_in[13];
    const float* final_w  = (const float*)d_in[14];
    const float* final_b  = (const float*)d_in[15];
    float*       out      = (float*)d_out;

    // f32 workspace layout
    float* ws   = (float*)d_ws;
    float* XH   = ws;                      // N*128  = 6,400,000 f
    float* ASRC = XH   + (size_t)NN * 128; // N*H
    float* ADST = ASRC + (size_t)NN * HH;
    float* EBUF = ADST + (size_t)NN * HH;  // E*H    = 3,200,000 f
    float* MBUF = EBUF + (size_t)EE * HH;  // N*H (u32 keys, then float m)
    float* DEN  = MBUF + (size_t)NN * HH;  // N*H
    float* COMB = DEN  + (size_t)NN * HH;  // N*192  = 9,600,000 f
    float* Qp   = COMB + (size_t)NN * 192; // N*64
    float* KP   = Qp   + (size_t)NN * 64;  // N*2*64
    float* VP   = KP   + (size_t)NN * KOBS * 64;
    float* WEND = VP   + (size_t)NN * KOBS * 64;

    // bf16 buffers: inputs overlay COMB (dead until memset after the GEMMs);
    // combined overlays XH (dead after edge aggregation); weights appended (tiny).
    bf16_t* Xb    = (bf16_t*)COMB;                  // N*128 bf16 (12.8 MB)
    bf16_t* Qfb   = Xb  + (size_t)NN * 128;         // N*128 bf16
    bf16_t* Obsb  = Qfb + (size_t)NN * 128;         // N*K*64 = N*128 bf16 (38.4 MB total = sizeof COMB)
    bf16_t* COMBb = (bf16_t*)XH;                    // N*192 bf16 (19.2 MB <= 25.6 MB)
    bf16_t* Wgb   = (bf16_t*)WEND;                  // 128*128
    bf16_t* Wqb   = Wgb + 128 * 128;                // 64*128
    bf16_t* Wkb   = Wqb + 64 * 128;                 // 64*64
    bf16_t* Wvb   = Wkb + 64 * 64;                  // 64*64
    bf16_t* Fwb   = Wvb + 64 * 64;                  // 64*192

    // 1) pack inputs + weights to bf16
    pack_bf16<<<pk_blocks(NN * 128), 256, 0, stream>>>(x,    Xb,   NN * 128);
    pack_bf16<<<pk_blocks(NN * 128), 256, 0, stream>>>(qf,   Qfb,  NN * 128);
    pack_bf16<<<pk_blocks(NN * 128), 256, 0, stream>>>(obsf, Obsb, NN * 128);
    pack_bf16<<<pk_blocks(128 * 128), 256, 0, stream>>>(gat_w,   Wgb, 128 * 128);
    pack_bf16<<<pk_blocks(64 * 128),  256, 0, stream>>>(wq,      Wqb, 64 * 128);
    pack_bf16<<<pk_blocks(64 * 64),   256, 0, stream>>>(wk,      Wkb, 64 * 64);
    pack_bf16<<<pk_blocks(64 * 64),   256, 0, stream>>>(wv,      Wvb, 64 * 64);
    pack_bf16<<<pk_blocks(64 * 192),  256, 0, stream>>>(final_w, Fwb, 64 * 192);

    // 2) dense projections on the WMMA path
    gemm_bf16<128, 128><<<NN / 16, 128, 0, stream>>>(Xb,   Wgb, XH, nullptr, 0);
    gemm_bf16<128,  64><<<NN / 16, 128, 0, stream>>>(Qfb,  Wqb, Qp, nullptr, 0);
    gemm_bf16< 64,  64><<<(NN * KOBS) / 16, 128, 0, stream>>>(Obsb, Wkb, KP, nullptr, 0);
    gemm_bf16< 64,  64><<<(NN * KOBS) / 16, 128, 0, stream>>>(Obsb, Wvb, VP, nullptr, 0);

    // 3) zero atomic accumulators {MBUF, DEN, COMB} (wipes Xb/Qfb/Obsb — now dead)
    hipMemsetAsync(MBUF, 0, (size_t)(NN * HH * 2 + NN * 192) * sizeof(float), stream);

    // 4) GAT attention pipeline
    int nh_blocks = (NN * HH + 255) / 256;
    int e_blocks  = (EE + 255) / 256;
    k_attn_coeff<<<nh_blocks, 256, 0, stream>>>(XH, att_src, att_dst, ASRC, ADST);
    k_edge_max<<<e_blocks, 256, 0, stream>>>(ei, ASRC, ADST, EBUF, (unsigned int*)MBUF);
    k_decode_max<<<nh_blocks, 256, 0, stream>>>((unsigned int*)MBUF);
    k_edge_exp<<<e_blocks, 256, 0, stream>>>(ei, EBUF, MBUF, DEN);
    k_edge_aggregate<<<(EE * 32) / 256, 256, 0, stream>>>(ei, EBUF, DEN, XH, COMB);

    // 5) obstacle attention epilogue + bias fold
    k_obstacle<<<nh_blocks, 256, 0, stream>>>(Qp, KP, VP, apos, opos, cdecay, gat_bias, COMB);

    // 6) pack combined to bf16 (overlays XH, dead now) and final fused projection
    pack_bf16<<<pk_blocks(NN * 192), 256, 0, stream>>>(COMB, COMBb, NN * 192);
    gemm_bf16<192, 64><<<NN / 16, 128, 0, stream>>>(COMBb, Fwb, out, final_b, 1);
}
